// DCGRUCell_16389595202009
// MI455X (gfx1250) — matrix operand order repaired and checked by reading, not compiled, not run on hardware
//
#include <hip/hip_runtime.h>
#include <hip/hip_bf16.h>

typedef __attribute__((ext_vector_type(16))) __bf16 v16bf;
typedef __attribute__((ext_vector_type(8)))  __bf16 v8bf;
typedef __attribute__((ext_vector_type(8)))  float  v8f;

constexpr int N_ = 8000;   // nodes
constexpr int U_ = 64;     // units
constexpr int D_ = 2;      // input dim
constexpr int E_ = 64000;  // edges per support
constexpr int B_ = 32;     // batch
constexpr int F_ = D_ + U_;     // 66
constexpr int M_ = 5;           // diffusion matrices
constexpr int FB = F_ * B_;     // 2112
constexpr int KC = F_ * M_;     // 330 (true K)
constexpr int KP = 352;         // K padded to 11*32
constexpr int KT = KP / 32;     // 11 k-tiles
constexpr int ROWS = B_ * N_;   // 256000 GEMM rows

// ---------------------------------------------------------------- build x0
// x0[n, f*B + b] = f<D ? inputs[b, n*D+f] : state[b, n*U+(f-D)]
__global__ __launch_bounds__(256) void build_x0(
    const float* __restrict__ inp, const float* __restrict__ st,
    float* __restrict__ x0) {
  long idx = (long)blockIdx.x * 256 + threadIdx.x;
  if (idx >= (long)N_ * FB) return;
  int n   = (int)(idx / FB);
  int rem = (int)(idx % FB);
  int f = rem >> 5;          // /32
  int b = rem & 31;
  float v = (f < D_) ? inp[(long)b * (N_ * D_) + n * D_ + f]
                     : st [(long)b * (N_ * U_) + n * U_ + (f - D_)];
  x0[idx] = v;
}

// ---------------------------------------------------------------- init
__global__ __launch_bounds__(256) void scale_init(
    const float* __restrict__ src, float alpha, float* __restrict__ dst, long n) {
  long i = (long)blockIdx.x * 256 + threadIdx.x;
  if (i < n) dst[i] = src ? alpha * src[i] : 0.0f;
}

// ---------------------------------------------------------------- SpMM
// y[row[e], :] += scale * val[e] * x[col[e], :]   (fp32 HW atomics)
__global__ __launch_bounds__(256) void spmm_atomic(
    const int* __restrict__ row, const int* __restrict__ col,
    const float* __restrict__ val, float scale,
    const float* __restrict__ x, float* __restrict__ y) {
  int e = blockIdx.x;
  int r = row[e];
  int c = col[e];
  float v = scale * val[e];
  const float* xs = x + (long)c * FB;
  float*       yd = y + (long)r * FB;
  for (int k = threadIdx.x; k < FB; k += 256)
    unsafeAtomicAdd(&yd[k], v * xs[k]);   // -> global_atomic_add_f32
}

// ---------------------------------------------------------------- pack xc
// xc_bf16[row, k], row = b*N+n, k = f*M+m (zeros for k>=330)
__global__ __launch_bounds__(256) void pack_xc(
    const float* __restrict__ xs_base, __bf16* __restrict__ xc) {
  long idx = (long)blockIdx.x * 256 + threadIdx.x;
  if (idx >= (long)ROWS * KP) return;
  long rowi = idx / KP;
  int  k    = (int)(idx % KP);
  int b = (int)(rowi / N_);
  int n = (int)(rowi % N_);
  float v = 0.0f;
  if (k < KC) {
    int f = k / M_;
    int m = k % M_;
    v = xs_base[(long)m * ((long)N_ * FB) + (long)n * FB + f * B_ + b];
  }
  xc[idx] = (__bf16)v;
}

// ---------------------------------------------------------------- pack W
// B-operand k order used here (contiguous, same as A) per (ktile, coltile):
//   lane<16 : col = lane,    K = kt*32 + j      (j=0..15)
//   lane>=16: col = lane-16, K = kt*32 + 16 + j
// stored so each lane reads 16 contiguous bf16 (32B).
__global__ __launch_bounds__(256) void pack_W(
    const float* __restrict__ W, int ncols, __bf16* __restrict__ Wpk) {
  int CT = ncols >> 4;
  long total = (long)KT * CT * 512;
  long idx = (long)blockIdx.x * 256 + threadIdx.x;
  if (idx >= total) return;
  int  j    = (int)(idx & 15);
  int  lane = (int)((idx >> 4) & 31);
  long tile = idx >> 9;
  int  ct = (int)(tile % CT);
  int  kt = (int)(tile / CT);
  int colI = ct * 16 + (lane & 15);
  int k    = kt * 32 + ((lane < 16) ? 0 : 16) + j;
  float v = (k < KC) ? W[(long)k * ncols + colI] : 0.0f;
  Wpk[idx] = (__bf16)v;
}

// ---------------------------------------------------------------- WMMA core
// One wave computes a 16-row tile across CT 16-col tiles, K-loop of 11.
template <int CT>
__device__ inline void wmma_row_tile(const __bf16* __restrict__ xc,
                                     const __bf16* __restrict__ Wpk,
                                     const float* __restrict__ bias,
                                     int rowTile, int lane, v8f acc[CT]) {
  int coll = lane & 15;
#pragma unroll
  for (int ct = 0; ct < CT; ++ct) {
    float bv = bias[ct * 16 + coll];
#pragma unroll
    for (int i = 0; i < 8; ++i) acc[ct][i] = bv;
  }
  // A k order used here (contiguous, matches Wpk): lanes 0-15 (M=lane): K 0-15; lanes 16-31: K 16-31
  int rowA = rowTile * 16 + coll;
  const __bf16* arow = xc + (long)rowA * KP;
  int half = (lane < 16) ? 0 : 16;
  for (int kt = 0; kt < KT; ++kt) {
    v8bf alo = *(const v8bf*)(arow + kt * 32 + half);
    v8bf ahi = *(const v8bf*)(arow + kt * 32 + 8 + half);
    v16bf a = __builtin_shufflevector(alo, ahi, 0, 1, 2, 3, 4, 5, 6, 7,
                                      8, 9, 10, 11, 12, 13, 14, 15);
#pragma unroll
    for (int ct = 0; ct < CT; ++ct) {
      v16bf bmat = *(const v16bf*)(Wpk + ((long)(kt * CT + ct) * 32 + lane) * 16);
      acc[ct] = __builtin_amdgcn_wmma_f32_16x16x32_bf16(
          false, a, false, bmat, (short)0, acc[ct], false, false);
    }
  }
}

// ---------------------------------------------------------------- GEMM (r,u)
// 128 cols; epilogue: sigmoid -> rh = r*hx (cols 0..63), u_buf (cols 64..127)
__global__ __launch_bounds__(256) void gemm_ru(
    const __bf16* __restrict__ xc, const __bf16* __restrict__ Wpk,
    const float* __restrict__ b_ru, const float* __restrict__ hx,
    float* __restrict__ rh, float* __restrict__ u_buf) {
  int wave = threadIdx.x >> 5;
  int lane = threadIdx.x & 31;
  int rowTile = blockIdx.x * 8 + wave;
  if (rowTile >= ROWS / 16) return;
  v8f acc[8];
  wmma_row_tile<8>(xc, Wpk, b_ru, rowTile, lane, acc);
  // D layout: vgpr i -> row tileBase + i + (lane>=16 ? 8:0), col = ct*16 + lane%16
  int rbase = rowTile * 16 + ((lane < 16) ? 0 : 8);
  int coll = lane & 15;
#pragma unroll
  for (int ct = 0; ct < 8; ++ct) {
    int colg = ct * 16 + coll;   // 0..127
#pragma unroll
    for (int i = 0; i < 8; ++i) {
      int rowg = rbase + i;
      int b = rowg / N_;
      int n = rowg % N_;
      float s = 1.0f / (1.0f + __expf(-acc[ct][i]));
      long hbase = (long)b * (N_ * U_) + (long)n * U_;
      if (colg < U_) rh[hbase + colg] = s * hx[hbase + colg];
      else           u_buf[hbase + (colg - U_)] = s;
    }
  }
}

// ---------------------------------------------------------------- GEMM (c)
// 64 cols; epilogue: out = u*hx + (1-u)*tanh(c)
__global__ __launch_bounds__(256) void gemm_c(
    const __bf16* __restrict__ xc, const __bf16* __restrict__ Wpk,
    const float* __restrict__ b_c, const float* __restrict__ hx,
    const float* __restrict__ u_buf, float* __restrict__ out) {
  int wave = threadIdx.x >> 5;
  int lane = threadIdx.x & 31;
  int rowTile = blockIdx.x * 8 + wave;
  if (rowTile >= ROWS / 16) return;
  v8f acc[4];
  wmma_row_tile<4>(xc, Wpk, b_c, rowTile, lane, acc);
  int rbase = rowTile * 16 + ((lane < 16) ? 0 : 8);
  int coll = lane & 15;
#pragma unroll
  for (int ct = 0; ct < 4; ++ct) {
    int colg = ct * 16 + coll;   // 0..63
#pragma unroll
    for (int i = 0; i < 8; ++i) {
      int rowg = rbase + i;
      int b = rowg / N_;
      int n = rowg % N_;
      float c = tanhf(acc[ct][i]);
      long hbase = (long)b * (N_ * U_) + (long)n * U_;
      float u = u_buf[hbase + colg];
      out[hbase + colg] = u * hx[hbase + colg] + (1.0f - u) * c;
    }
  }
}

// ================================================================ host side
extern "C" void kernel_launch(void* const* d_in, const int* in_sizes, int n_in,
                              void* d_out, int out_size, void* d_ws, size_t ws_size,
                              hipStream_t stream) {
  (void)in_sizes; (void)n_in; (void)out_size; (void)ws_size;
  const float* inputs = (const float*)d_in[0];
  const float* hx     = (const float*)d_in[1];
  const int*   row0   = (const int*)  d_in[2];
  const int*   col0   = (const int*)  d_in[3];
  const float* val0   = (const float*)d_in[4];
  const int*   row1   = (const int*)  d_in[5];
  const int*   col1   = (const int*)  d_in[6];
  const float* val1   = (const float*)d_in[7];
  const float* W_ru   = (const float*)d_in[8];
  const float* b_ru   = (const float*)d_in[9];
  const float* W_c    = (const float*)d_in[10];
  const float* b_c    = (const float*)d_in[11];
  float* out = (float*)d_out;

  // ---- workspace carve-up
  const long nX0 = (long)N_ * FB;               // 16,896,000 floats per matrix
  char* p = (char*)d_ws;
  float*  xs     = (float*)p;  p += 5 * nX0 * sizeof(float);
  __bf16* xc     = (__bf16*)p; p += (long)ROWS * KP * sizeof(__bf16);
  __bf16* Wru_pk = (__bf16*)p; p += (long)KT * 8 * 512 * sizeof(__bf16);
  __bf16* Wc_pk  = (__bf16*)p; p += (long)KT * 4 * 512 * sizeof(__bf16);
  float*  u_buf  = (float*)p;  p += (long)B_ * N_ * U_ * sizeof(float);
  float*  rh     = (float*)p;  p += (long)B_ * N_ * U_ * sizeof(float);

  const int g_x0 = (int)((nX0 + 255) / 256);
  const long nXC = (long)ROWS * KP;
  const int g_xc = (int)((nXC + 255) / 256);

  // ---- weight pre-swizzle into WMMA B-operand layout (tiny, once per call)
  pack_W<<<(KT * 8 * 512 + 255) / 256, 256, 0, stream>>>(W_ru, 128, Wru_pk);
  pack_W<<<(KT * 4 * 512 + 255) / 256, 256, 0, stream>>>(W_c, 64, Wc_pk);

  // ---- graph conv front half (x0 + Chebyshev diffusion + bf16 pack)
  auto gconv_front = [&](const float* state) {
    build_x0<<<g_x0, 256, 0, stream>>>(inputs, state, xs);
    for (int s = 0; s < 2; ++s) {
      const int*   rw = s ? row1 : row0;
      const int*   cl = s ? col1 : col0;
      const float* vl = s ? val1 : val0;
      float* x1 = xs + (long)(1 + 2 * s) * nX0;
      float* x2 = xs + (long)(2 + 2 * s) * nX0;
      scale_init<<<g_x0, 256, 0, stream>>>(nullptr, 0.0f, x1, nX0);
      spmm_atomic<<<E_, 256, 0, stream>>>(rw, cl, vl, 1.0f, xs, x1);
      scale_init<<<g_x0, 256, 0, stream>>>(xs, -1.0f, x2, nX0);      // x2 = -x0
      spmm_atomic<<<E_, 256, 0, stream>>>(rw, cl, vl, 2.0f, x1, x2); // += 2*A*x1
    }
    pack_xc<<<g_xc, 256, 0, stream>>>(xs, xc);
  };

  // ---- gconv #1 -> gates (r,u)
  gconv_front(hx);
  gemm_ru<<<ROWS / 16 / 8, 256, 0, stream>>>(xc, Wru_pk, b_ru, hx, rh, u_buf);

  // ---- gconv #2 (state = r*hx) -> candidate c, fused final combine
  gconv_front(rh);
  gemm_c<<<ROWS / 16 / 8, 256, 0, stream>>>(xc, Wc_pk, b_c, hx, u_buf, out);
}
